// Predictor_10849087390113
// MI455X (gfx1250) — matrix operand order repaired and checked by reading, not compiled, not run on hardware
//
#include <hip/hip_runtime.h>
#include <hip/hip_bf16.h>
#include <stdint.h>

// Problem constants (from reference)
#define BB   8
#define NN   1024
#define DIN_ 768
#define DD   384
#define LL   6
#define HH   12
#define HD_  32
#define BN   (BB * NN)
#define D3   (3 * DD)
#define D4   (4 * DD)

typedef _Float16 f16;
typedef f16  v16h __attribute__((ext_vector_type(16)));
typedef f16  v8h  __attribute__((ext_vector_type(8)));
typedef f16  v4h  __attribute__((ext_vector_type(4)));
typedef float v8f __attribute__((ext_vector_type(8)));

union H16 { v16h v; v8h h[2]; };

static __device__ __forceinline__ v8f wmma16(const H16& a, const H16& b, v8f c) {
  return __builtin_amdgcn_wmma_f32_16x16x32_f16(false, a.v, false, b.v,
                                                (short)0, c, false, false);
}

// ---------------------------------------------------------------------------
// f32 -> f16 conversion (grid-stride)
// ---------------------------------------------------------------------------
__global__ void cvt_f16_kernel(const float* __restrict__ in, f16* __restrict__ out, int n) {
  int i = blockIdx.x * blockDim.x + threadIdx.x;
  int stride = gridDim.x * blockDim.x;
  for (; i < n; i += stride) out[i] = (f16)in[i];
}

// ---------------------------------------------------------------------------
// Embed epilogue: mask-token substitution + positional embeddings (in place)
// ---------------------------------------------------------------------------
__global__ void embed_fuse_kernel(float* __restrict__ h,
                                  const unsigned char* __restrict__ tmask,
                                  const int* __restrict__ hid,
                                  const int* __restrict__ wid,
                                  const float* __restrict__ mtok,
                                  const float* __restrict__ he,
                                  const float* __restrict__ we) {
  int row = blockIdx.x;        // 0..BN-1
  int col = threadIdx.x;       // 0..DD-1 (blockDim.x == 384)
  size_t idx = (size_t)row * DD + col;
  float v = tmask[row] ? mtok[col] : h[idx];
  v += he[(size_t)hid[row] * DD + col] + we[(size_t)wid[row] * DD + col];
  h[idx] = v;
}

// ---------------------------------------------------------------------------
// LayerNorm over D=384, f32 in -> f16 out. One block (128 threads) per row.
// ---------------------------------------------------------------------------
__global__ __launch_bounds__(128)
void layernorm_f16_kernel(const float* __restrict__ x,
                          const float* __restrict__ w,
                          const float* __restrict__ b,
                          f16* __restrict__ out) {
  __shared__ float s1[128];
  __shared__ float s2[128];
  int row = blockIdx.x, t = threadIdx.x;
  const float* xr = x + (size_t)row * DD;
  float v0 = xr[t], v1 = xr[t + 128], v2 = xr[t + 256];
  s1[t] = v0 + v1 + v2;
  s2[t] = v0 * v0 + v1 * v1 + v2 * v2;
  __syncthreads();
  for (int s = 64; s > 0; s >>= 1) {
    if (t < s) { s1[t] += s1[t + s]; s2[t] += s2[t + s]; }
    __syncthreads();
  }
  float mean = s1[0] * (1.0f / DD);
  float var  = s2[0] * (1.0f / DD) - mean * mean;
  float rinv = rsqrtf(var + 1e-5f);
  f16* o = out + (size_t)row * DD;
  o[t]       = (f16)((v0 - mean) * rinv * w[t]       + b[t]);
  o[t + 128] = (f16)((v1 - mean) * rinv * w[t + 128] + b[t + 128]);
  o[t + 256] = (f16)((v2 - mean) * rinv * w[t + 256] + b[t + 256]);
}

// ---------------------------------------------------------------------------
// Tiled WMMA GEMM:  OUT[M,Nc] = A[M,K](f16) @ W[K,Nc](f16) + bias
// Block tile 128x128, K-step 32, 256 threads = 8 waves (4x2), wave tile 32x64.
// Double-buffered LDS (ping-pong), one barrier per K-step; next-tile global
// loads are issued before the WMMA block so load latency overlaps math.
// EPI: 0 = store f32, 1 = store f16, 2 = GELU + store f16,
//      3 = residual: outf += ls[col] * (acc + bias)   (Nc == DD)
// ---------------------------------------------------------------------------
template <int EPI>
__global__ __launch_bounds__(256)
void gemm_wmma_kernel(const f16* __restrict__ A, const f16* __restrict__ W,
                      const float* __restrict__ bias, const float* __restrict__ ls,
                      float* __restrict__ outf, f16* __restrict__ outh,
                      int M, int K, int Nc) {
  __shared__ __align__(16) f16 As[2][128][32];
  __shared__ __align__(16) f16 Bt[2][128][32];   // W tile transposed: Bt[.][col][k]

  const int tid  = threadIdx.x;
  const int lane = tid & 31;
  const int hi   = lane >> 4;        // 0 / 1 half-wave
  const int l16  = lane & 15;
  const int wid  = tid >> 5;         // 0..7
  const int wm   = wid >> 1;         // 0..3 (row tiles of 32)
  const int wn   = wid & 1;          // 0..1 (col tiles of 64)
  const int m0   = blockIdx.y * 128;
  const int n0   = blockIdx.x * 128;

  // staging assignments
  const int arow = tid >> 2;          // 0..63  (this row and +64)
  const int ach  = (tid & 3) * 8;     // 0,8,16,24
  const int wkk  = tid >> 3;          // 0..31
  const int wcc  = (tid & 7) * 16;    // 0..112

  v8f acc[2][4] = {};

  // ---- preload K-tile 0 into buffer 0 ----
  {
    *(v8h*)&As[0][arow][ach]      = *(const v8h*)&A[(size_t)(m0 + arow) * K + ach];
    *(v8h*)&As[0][arow + 64][ach] = *(const v8h*)&A[(size_t)(m0 + arow + 64) * K + ach];
    v8h w0 = *(const v8h*)&W[(size_t)wkk * Nc + n0 + wcc];
    v8h w1 = *(const v8h*)&W[(size_t)wkk * Nc + n0 + wcc + 8];
#pragma unroll
    for (int j = 0; j < 8; ++j) {
      Bt[0][wcc + j][wkk]     = w0[j];
      Bt[0][wcc + 8 + j][wkk] = w1[j];
    }
  }
  __syncthreads();

  const int KT = K / 32;
  for (int kt = 0; kt < KT; ++kt) {
    const int cur = kt & 1, nxt = cur ^ 1;
    const bool has_next = (kt + 1 < KT);
    v8h a0, a1, w0, w1;
    if (has_next) {
      int k0 = (kt + 1) * 32;
      a0 = *(const v8h*)&A[(size_t)(m0 + arow) * K + k0 + ach];
      a1 = *(const v8h*)&A[(size_t)(m0 + arow + 64) * K + k0 + ach];
      w0 = *(const v8h*)&W[(size_t)(k0 + wkk) * Nc + n0 + wcc];
      w1 = *(const v8h*)&W[(size_t)(k0 + wkk) * Nc + n0 + wcc + 8];
      if (kt + 2 < KT)
        __builtin_prefetch(&A[(size_t)(m0 + arow) * K + (kt + 2) * 32 + ach], 0, 1);
    }

    // ---- fragments (ISA wave32 layouts) + 8 WMMAs ----
    H16 af[2], bf[4];
#pragma unroll
    for (int mi = 0; mi < 2; ++mi) {
      int r  = wm * 32 + mi * 16 + l16;
      int kb = hi * 8;
      af[mi].h[0] = *(const v8h*)&As[cur][r][kb];
      af[mi].h[1] = *(const v8h*)&As[cur][r][kb + 16];
    }
#pragma unroll
    for (int ni = 0; ni < 4; ++ni) {
      int c  = wn * 64 + ni * 16 + l16;
      int kb = hi * 8;
      bf[ni].h[0] = *(const v8h*)&Bt[cur][c][kb];
      bf[ni].h[1] = *(const v8h*)&Bt[cur][c][kb + 16];
    }
#pragma unroll
    for (int mi = 0; mi < 2; ++mi)
#pragma unroll
      for (int ni = 0; ni < 4; ++ni)
        acc[mi][ni] = wmma16(af[mi], bf[ni], acc[mi][ni]);

    // ---- stage next tile into the other buffer ----
    if (has_next) {
      *(v8h*)&As[nxt][arow][ach]      = a0;
      *(v8h*)&As[nxt][arow + 64][ach] = a1;
#pragma unroll
      for (int j = 0; j < 8; ++j) {
        Bt[nxt][wcc + j][wkk]     = w0[j];
        Bt[nxt][wcc + 8 + j][wkk] = w1[j];
      }
    }
    __syncthreads();
  }

  // ---- epilogue ----
#pragma unroll
  for (int mi = 0; mi < 2; ++mi) {
#pragma unroll
    for (int ni = 0; ni < 4; ++ni) {
      int col  = n0 + wn * 64 + ni * 16 + l16;
      int rowb = m0 + wm * 32 + mi * 16 + hi * 8;
      float bv  = bias[col];
      float lsv = (EPI == 3) ? ls[col] : 0.0f;
#pragma unroll
      for (int r = 0; r < 8; ++r) {
        float v = acc[mi][ni][r] + bv;
        size_t idx = (size_t)(rowb + r) * Nc + col;
        if (EPI == 0) {
          outf[idx] = v;
        } else if (EPI == 1) {
          outh[idx] = (f16)v;
        } else if (EPI == 2) {
          v = 0.5f * v * (1.0f + erff(v * 0.70710678118654752f));  // exact GELU
          outh[idx] = (f16)v;
        } else {
          outf[idx] += lsv * v;
        }
      }
    }
  }
}

// ---------------------------------------------------------------------------
// Flash-style attention with WMMA, double-buffered K/V staging.
// Grid: (NN/128, BB*HH). Block: 256 threads = 8 waves; wave = 16 query rows.
// qkv layout: [(b*NN + n) * 3D + part*D + h*HD + d], f16.
// ---------------------------------------------------------------------------
__global__ __launch_bounds__(256)
void attention_wmma_kernel(const f16* __restrict__ qkv,
                           const int* __restrict__ sid,
                           const unsigned char* __restrict__ smask,
                           f16* __restrict__ attno) {
  __shared__ __align__(16) f16 Ks[2][32][32];     // [buf][key][dim]
  __shared__ __align__(16) f16 Vt[2][32][32];     // [buf][dim][key]
  __shared__ __align__(16) f16 Ps[8][16][32];     // per-wave P staging [row][key]

  const int tid  = threadIdx.x;
  const int lane = tid & 31;
  const int hi   = lane >> 4;
  const int l16  = lane & 15;
  const int wvid = tid >> 5;
  const int b    = blockIdx.y / HH;
  const int hh   = blockIdx.y % HH;
  const int q0   = blockIdx.x * 128 + wvid * 16;

  const float scale = 0.17677669529663687f;  // 1/sqrt(HD)
  const float MINF  = -3.4028235e38f;        // finfo(f32).min, as in reference

  // staging assignment for K/V chunks
  const int key = tid >> 3;         // 0..31
  const int cc  = (tid & 7) * 4;    // 0..28

  // Q fragment: A-layout, loaded straight from global (HD == 32 == WMMA K)
  H16 qf;
  {
    int qr = q0 + l16;
    const f16* qp = qkv + (size_t)(b * NN + qr) * D3 + hh * HD_;
    int kb = hi * 8;
    qf.h[0] = *(const v8h*)&qp[kb];
    qf.h[1] = *(const v8h*)&qp[kb + 16];
  }

  // Per-lane metadata for its 8 C/D-fragment rows (M = r + 8*hi)
  int   qsid[8];
  float qsm[8];
#pragma unroll
  for (int r = 0; r < 8; ++r) {
    int qr  = q0 + r + 8 * hi;
    qsid[r] = sid[b * NN + qr];
    qsm[r]  = smask[b * NN + qr] ? 1.0f : 0.0f;
  }

  float m_[8], l_[8];
#pragma unroll
  for (int r = 0; r < 8; ++r) { m_[r] = MINF; l_[r] = 0.0f; }
  v8f O0 = {}, O1 = {};
  const v8f zacc = {};

  // ---- preload chunk 0 ----
  {
    size_t base = (size_t)(b * NN + key) * D3 + hh * HD_ + cc;
    *(v4h*)&Ks[0][key][cc] = *(const v4h*)&qkv[base + DD];
    v4h vv = *(const v4h*)&qkv[base + 2 * DD];
#pragma unroll
    for (int j = 0; j < 4; ++j) Vt[0][cc + j][key] = vv[j];
  }
  __syncthreads();

  const int JT = NN / 32;
  for (int jt = 0; jt < JT; ++jt) {
    const int j0 = jt * 32;
    const int cur = jt & 1, nxt = cur ^ 1;
    const bool has_next = (jt + 1 < JT);

    // issue next chunk's global loads before the math
    v4h kreg, vreg;
    if (has_next) {
      size_t base = (size_t)(b * NN + j0 + 32 + key) * D3 + hh * HD_ + cc;
      kreg = *(const v4h*)&qkv[base + DD];
      vreg = *(const v4h*)&qkv[base + 2 * DD];
    }

    // --- S = Q @ K^T (two 16-key halves) ---
    H16 kf0, kf1;
    {
      int kb = hi * 8;
      kf0.h[0] = *(const v8h*)&Ks[cur][l16][kb];
      kf0.h[1] = *(const v8h*)&Ks[cur][l16][kb + 16];
      kf1.h[0] = *(const v8h*)&Ks[cur][16 + l16][kb];
      kf1.h[1] = *(const v8h*)&Ks[cur][16 + l16][kb + 16];
    }
    v8f S0 = wmma16(qf, kf0, zacc);
    v8f S1 = wmma16(qf, kf1, zacc);

    // --- mask + scale (column key for this lane) ---
    int   ks0 = sid[b * NN + j0 + l16];
    int   ks1 = sid[b * NN + j0 + 16 + l16];
    float km0 = smask[b * NN + j0 + l16] ? 1.0f : 0.0f;
    float km1 = smask[b * NN + j0 + 16 + l16] ? 1.0f : 0.0f;

    float p0[8], p1[8];
#pragma unroll
    for (int r = 0; r < 8; ++r) {
      bool mk0 = (ks0 != qsid[r]) | (qsm[r] != 0.0f) | (km0 != 0.0f);
      bool mk1 = (ks1 != qsid[r]) | (qsm[r] != 0.0f) | (km1 != 0.0f);
      p0[r] = mk0 ? MINF : S0[r] * scale;
      p1[r] = mk1 ? MINF : S1[r] * scale;
    }

    // --- online softmax (row reductions across the 16-lane half-wave) ---
#pragma unroll
    for (int r = 0; r < 8; ++r) {
      float mx = fmaxf(p0[r], p1[r]);
#pragma unroll
      for (int off = 1; off < 16; off <<= 1) mx = fmaxf(mx, __shfl_xor(mx, off, 16));
      float mnew = fmaxf(m_[r], mx);
      float sc = __expf(m_[r] - mnew);
      float e0 = __expf(p0[r] - mnew);
      float e1 = __expf(p1[r] - mnew);
      float rs = e0 + e1;
#pragma unroll
      for (int off = 1; off < 16; off <<= 1) rs += __shfl_xor(rs, off, 16);
      l_[r] = l_[r] * sc + rs;
      m_[r] = mnew;
      O0[r] *= sc;
      O1[r] *= sc;
      p0[r] = e0;
      p1[r] = e1;
    }

    // --- P (C-layout) -> A-layout via per-wave LDS staging ---
#pragma unroll
    for (int r = 0; r < 8; ++r) {
      Ps[wvid][r + 8 * hi][l16]      = (f16)p0[r];
      Ps[wvid][r + 8 * hi][16 + l16] = (f16)p1[r];
    }
    __builtin_amdgcn_wave_barrier();
    H16 pf;
    {
      int kb = hi * 8;
      pf.h[0] = *(const v8h*)&Ps[wvid][l16][kb];
      pf.h[1] = *(const v8h*)&Ps[wvid][l16][kb + 16];
    }

    // --- O += P @ V (two 16-dim halves) ---
    H16 vf0, vf1;
    {
      int kb = hi * 8;
      vf0.h[0] = *(const v8h*)&Vt[cur][l16][kb];
      vf0.h[1] = *(const v8h*)&Vt[cur][l16][kb + 16];
      vf1.h[0] = *(const v8h*)&Vt[cur][16 + l16][kb];
      vf1.h[1] = *(const v8h*)&Vt[cur][16 + l16][kb + 16];
    }
    O0 = wmma16(pf, vf0, O0);
    O1 = wmma16(pf, vf1, O1);

    // --- stage next chunk into the other buffer ---
    if (has_next) {
      *(v4h*)&Ks[nxt][key][cc] = kreg;
#pragma unroll
      for (int j = 0; j < 4; ++j) Vt[nxt][cc + j][key] = vreg[j];
    }
    __syncthreads();
  }

  // --- normalize by l and store f16 ---
#pragma unroll
  for (int r = 0; r < 8; ++r) {
    float rl = 1.0f / l_[r];
    int qr = q0 + r + 8 * hi;
    f16* op = attno + (size_t)(b * NN + qr) * DD + hh * HD_;
    op[l16]      = (f16)(O0[r] * rl);
    op[16 + l16] = (f16)(O1[r] * rl);
  }
}

// ---------------------------------------------------------------------------
// Host-side orchestration
// ---------------------------------------------------------------------------
extern "C" void kernel_launch(void* const* d_in, const int* in_sizes, int n_in,
                              void* d_out, int out_size, void* d_ws, size_t ws_size,
                              hipStream_t stream) {
  const float* x        = (const float*)d_in[0];
  const int*   hid      = (const int*)d_in[1];
  const int*   wid      = (const int*)d_in[2];
  const unsigned char* tmask = (const unsigned char*)d_in[3];
  const int*   sid      = (const int*)d_in[4];
  const unsigned char* smask = (const unsigned char*)d_in[5];
  // d_in[6] == r, unused by the reference
  const float* embed_w  = (const float*)d_in[7];
  const float* embed_b  = (const float*)d_in[8];
  const float* h_embed  = (const float*)d_in[9];
  const float* w_embed  = (const float*)d_in[10];
  const float* mask_tok = (const float*)d_in[11];
  const float* norm1_w  = (const float*)d_in[12];
  const float* norm1_b  = (const float*)d_in[13];
  const float* ls1      = (const float*)d_in[14];
  const float* qkv_w    = (const float*)d_in[15];
  const float* qkv_b    = (const float*)d_in[16];
  const float* proj_w   = (const float*)d_in[17];
  const float* proj_b   = (const float*)d_in[18];
  const float* norm2_w  = (const float*)d_in[19];
  const float* norm2_b  = (const float*)d_in[20];
  const float* fc1_w    = (const float*)d_in[21];
  const float* fc1_b    = (const float*)d_in[22];
  const float* fc2_w    = (const float*)d_in[23];
  const float* fc2_b    = (const float*)d_in[24];
  const float* ls2      = (const float*)d_in[25];
  const float* pnorm_w  = (const float*)d_in[26];
  const float* pnorm_b  = (const float*)d_in[27];
  const float* pproj_w  = (const float*)d_in[28];
  const float* pproj_b  = (const float*)d_in[29];

  char* ws = (char*)d_ws;
  size_t off = 0;
  auto alloc = [&](size_t bytes) -> void* {
    void* p = ws + off;
    off = (off + bytes + 255) & ~(size_t)255;
    return p;
  };
  f16*   xf     = (f16*)alloc((size_t)BN * DIN_ * 2);
  float* h      = (float*)alloc((size_t)BN * DD * 4);
  f16*   hn     = (f16*)alloc((size_t)BN * DD * 2);
  f16*   qkvb   = (f16*)alloc((size_t)BN * D3 * 2);
  f16*   attno  = (f16*)alloc((size_t)BN * DD * 2);
  f16*   mlp1   = (f16*)alloc((size_t)BN * D4 * 2);
  f16*   wEmbed = (f16*)alloc((size_t)DIN_ * DD * 2);
  f16*   wQkv   = (f16*)alloc((size_t)LL * DD * D3 * 2);
  f16*   wProj  = (f16*)alloc((size_t)LL * DD * DD * 2);
  f16*   wFc1   = (f16*)alloc((size_t)LL * DD * D4 * 2);
  f16*   wFc2   = (f16*)alloc((size_t)LL * D4 * DD * 2);
  f16*   wPproj = (f16*)alloc((size_t)DD * DIN_ * 2);
  (void)in_sizes; (void)n_in; (void)out_size; (void)ws_size;

  auto cvt = [&](const float* src, f16* dst, size_t n) {
    cvt_f16_kernel<<<1024, 256, 0, stream>>>(src, dst, (int)n);
  };
  cvt(x,       xf,     (size_t)BN * DIN_);
  cvt(embed_w, wEmbed, (size_t)DIN_ * DD);
  cvt(qkv_w,   wQkv,   (size_t)LL * DD * D3);
  cvt(proj_w,  wProj,  (size_t)LL * DD * DD);
  cvt(fc1_w,   wFc1,   (size_t)LL * DD * D4);
  cvt(fc2_w,   wFc2,   (size_t)LL * D4 * DD);
  cvt(pproj_w, wPproj, (size_t)DD * DIN_);

  // Embed: h = x @ embed_w + embed_b   (f32 out), then mask/pos-embed fuse
  gemm_wmma_kernel<0><<<dim3(DD / 128, BN / 128), 256, 0, stream>>>(
      xf, wEmbed, embed_b, nullptr, h, nullptr, BN, DIN_, DD);
  embed_fuse_kernel<<<BN, DD, 0, stream>>>(h, tmask, hid, wid, mask_tok, h_embed, w_embed);

  for (int i = 0; i < LL; ++i) {
    // Attention branch
    layernorm_f16_kernel<<<BN, 128, 0, stream>>>(h, norm1_w + i * DD, norm1_b + i * DD, hn);
    gemm_wmma_kernel<1><<<dim3(D3 / 128, BN / 128), 256, 0, stream>>>(
        hn, wQkv + (size_t)i * DD * D3, qkv_b + i * D3, nullptr, nullptr, qkvb, BN, DD, D3);
    attention_wmma_kernel<<<dim3(NN / 128, BB * HH), 256, 0, stream>>>(qkvb, sid, smask, attno);
    gemm_wmma_kernel<3><<<dim3(DD / 128, BN / 128), 256, 0, stream>>>(
        attno, wProj + (size_t)i * DD * DD, proj_b + i * DD, ls1 + i * DD, h, nullptr, BN, DD, DD);
    // MLP branch
    layernorm_f16_kernel<<<BN, 128, 0, stream>>>(h, norm2_w + i * DD, norm2_b + i * DD, hn);
    gemm_wmma_kernel<2><<<dim3(D4 / 128, BN / 128), 256, 0, stream>>>(
        hn, wFc1 + (size_t)i * DD * D4, fc1_b + i * D4, nullptr, nullptr, mlp1, BN, DD, D4);
    gemm_wmma_kernel<3><<<dim3(DD / 128, BN / 128), 256, 0, stream>>>(
        mlp1, wFc2 + (size_t)i * D4 * DD, fc2_b + i * DD, ls2 + i * DD, h, nullptr, BN, D4, DD);
  }

  // Final norm + projection to DIN (f32 out -> d_out)
  layernorm_f16_kernel<<<BN, 128, 0, stream>>>(h, pnorm_w, pnorm_b, hn);
  gemm_wmma_kernel<0><<<dim3(DIN_ / 128, BN / 128), 256, 0, stream>>>(
      hn, wPproj, pproj_b, nullptr, (float*)d_out, nullptr, BN, DD, DIN_);
}
